// AttentionAggregator2_46033459479180
// MI455X (gfx1250) — matrix operand order repaired and checked by reading, not compiled, not run on hardware
//
#include <hip/hip_runtime.h>
#include <hip/hip_bf16.h>

// ---------------------------------------------------------------------------
// Problem constants (from reference)
// ---------------------------------------------------------------------------
#define N_NODES 16384
#define K_NB    32
#define D_IN    256
#define E_DIM   64
#define DE      320      // D + E
#define H_DIM   64
#define O_DIM   256

// CDNA5 WMMA types
typedef __attribute__((ext_vector_type(16))) __bf16 v16bf;
typedef __attribute__((ext_vector_type(8)))  float  v8f;

#define WMMA_BF16(a, b, c) \
  __builtin_amdgcn_wmma_f32_16x16x32_bf16(false, (a), false, (b), (short)0, (c), false, false)

// LDS row strides (bf16 elements). Multiples of 8 so every 16B access is aligned.
#define SROW 328   // 320 + 8 pad : neibs_cat / att2_w1
#define SH   72    // 64 + 8 pad  : h1 / att2_w2

// ---------------------------------------------------------------------------
// Fragment loads per ISA 7.12.2 layouts (wave32, 16-bit A 16x32 / B 32x16)
// ---------------------------------------------------------------------------
__device__ __forceinline__ v16bf load_a_frag(const __bf16* base, int stride,
                                             int row, int kb, int lh) {
  // lane<16: K = kb+0..7 then kb+16..23 ; lane>=16: K = kb+8..15 then kb+24..31
  v16bf f;
  const __bf16* p = base + row * stride + kb + lh * 8;
  ((uint4*)&f)[0] = *(const uint4*)p;
  ((uint4*)&f)[1] = *(const uint4*)(p + 16);
  return f;
}

__device__ __forceinline__ v16bf load_b_frag(const __bf16* w, int stride,
                                             int ncol, int kb, int lh) {
  // B[k][n] = W[n][k] (W row-major [n][k]); lane<16: K=kb+0..7 then kb+16..23, lane>=16: K=kb+8..15 then kb+24..31
  v16bf f;
  const __bf16* p = w + ncol * stride + kb + lh * 8;
  ((uint4*)&f)[0] = *(const uint4*)p;
  ((uint4*)&f)[1] = *(const uint4*)(p + 16);
  return f;
}

__device__ __forceinline__ void st4bf(__bf16* p, float4 v) {
  __bf16 tmp[4] = {(__bf16)v.x, (__bf16)v.y, (__bf16)v.z, (__bf16)v.w};
  *(uint2*)p = *(uint2*)tmp;   // single 8B LDS store
}

// CDNA5 async global->LDS copy (ASYNCcnt-tracked, no VGPR data round-trip).
// Generic LDS pointers carry the LDS byte offset in their low 32 bits
// (ISA 10.2 aperture mapping).
__device__ __forceinline__ void async_copy_b128(void* lds_ptr, const void* gptr) {
  unsigned lds_off = (unsigned)(unsigned long long)lds_ptr;
  asm volatile("global_load_async_to_lds_b128 %0, %1, off"
               :: "v"(lds_off), "v"(gptr)
               : "memory");
}
__device__ __forceinline__ void wait_asynccnt0() {
  asm volatile("s_wait_asynccnt 0x0" ::: "memory");
}

// ---------------------------------------------------------------------------
// Kernel 1: fused attention. One WG = 4 nodes = 128 neighbor rows.
// ~157 KB LDS -> 2 workgroups per 320KB WGP so one WG's WMMA phase overlaps
// the other's HBM stream. neibs_cat streamed once HBM -> LDS(bf16); both
// attention GEMMs on WMMA; ws computed straight from GEMM2 accumulators via
// cross-lane reduction (no LDS round-trip). Produces agg[N][320] fp32 scratch.
// ---------------------------------------------------------------------------
__global__ __launch_bounds__(256)
void attn_fused_kernel(const float* __restrict__ x,
                       const float* __restrict__ neibs,
                       const float* __restrict__ edge,
                       const float* __restrict__ mask,
                       const float* __restrict__ att_w1,
                       const float* __restrict__ att_w2,
                       const float* __restrict__ att2_w1,
                       const float* __restrict__ att2_w2,
                       float* __restrict__ agg_out) {
  extern __shared__ __align__(16) char smem_raw[];
  __bf16* sm_cat  = (__bf16*)smem_raw;           // [128][SROW]
  __bf16* sm_w1   = sm_cat + 128 * SROW;         // [64][SROW]  att2_w1
  __bf16* sm_w2   = sm_w1 + 64 * SROW;           // [64][SH]    att2_w2
  __bf16* sm_h1   = sm_w2 + 64 * SH;             // [128][SH]   tanh(cat@w1.T)
  float*  sm_x    = (float*)(sm_h1 + 128 * SH);  // [4][256]
  float*  sm_hx   = sm_x + 4 * 256;              // [4][64]
  float*  sm_xatt = sm_hx + 4 * 64;              // [4][64]
  float*  sm_ws   = sm_xatt + 4 * 64;            // [4][32]
  float*  sm_mask = sm_ws + 4 * 32;              // [4][32]

  const int t     = threadIdx.x;
  const int node0 = blockIdx.x * 4;
  const long row0 = (long)node0 * K_NB;

  const int wave = t >> 5;
  const int lane = t & 31;
  const int lr   = lane & 15;
  const int lh   = lane >> 4;

  // ---- Async global->LDS for fp32 data that needs no conversion -----------
  // x rows: 4x256 fp32 = 256 b128 chunks, exactly one per thread.
  {
    int r = t >> 6, c4 = t & 63;
    async_copy_b128(sm_x + t * 4, x + (long)(node0 + r) * D_IN + c4 * 4);
  }
  if (t < 32) {  // mask: 4x32 fp32 = 32 b128 chunks
    async_copy_b128(sm_mask + t * 4, mask + row0 + t * 4);
  }

  // ---- Stage neibs_cat (fp32 -> bf16) and attention weights ---------------
  for (int i = t; i < 128 * 64; i += 256) {              // neibs: 128 rows x 64 float4
    int r = i >> 6, c4 = i & 63;
    float4 v = ((const float4*)(neibs + (row0 + r) * D_IN))[c4];
    st4bf(sm_cat + r * SROW + c4 * 4, v);
  }
  for (int i = t; i < 128 * 16; i += 256) {              // edge: 128 rows x 16 float4
    int r = i >> 4, c4 = i & 15;
    float4 v = ((const float4*)(edge + (row0 + r) * E_DIM))[c4];
    st4bf(sm_cat + r * SROW + D_IN + c4 * 4, v);
  }
  for (int i = t; i < 64 * 80; i += 256) {               // att2_w1: 64 x 80 float4
    int r = i / 80, c4 = i % 80;
    float4 v = ((const float4*)(att2_w1 + (long)r * DE))[c4];
    st4bf(sm_w1 + r * SROW + c4 * 4, v);
  }
  for (int i = t; i < 64 * 16; i += 256) {               // att2_w2: 64 x 16 float4
    int r = i >> 4, c4 = i & 15;
    float4 v = ((const float4*)(att2_w2 + (long)r * H_DIM))[c4];
    st4bf(sm_w2 + r * SH + c4 * 4, v);
  }
  wait_asynccnt0();          // drain this wave's async copies before signaling
  __syncthreads();

  // ---- x_att (M=4: VALU is cheapest) --------------------------------------
  {
    int n = t >> 6, h = t & 63;                          // 256 dots, 1/thread
    const float* wr = att_w1 + (long)h * D_IN;
    const float* xr = sm_x + n * 256;
    float acc = 0.f;
#pragma unroll 8
    for (int d = 0; d < D_IN; ++d) acc += wr[d] * xr[d];
    sm_hx[n * 64 + h] = tanhf(acc);
  }
  __syncthreads();
  {
    int n = t >> 6, h = t & 63;
    const float* wr = att_w2 + (long)h * H_DIM;
    const float* hr = sm_hx + n * 64;
    float acc = 0.f;
#pragma unroll 8
    for (int d = 0; d < H_DIM; ++d) acc += wr[d] * hr[d];
    sm_xatt[n * 64 + h] = acc;
  }
  __syncthreads();

  // ---- GEMM1: h1 = tanh(cat[128x320] @ att2_w1.T[320x64]) -----------------
  // wave w owns M-tile rows [w*16, w*16+16) x 4 N-tiles
  const int mrow0 = wave * 16;
  {
    v8f acc[4];
#pragma unroll
    for (int n = 0; n < 4; ++n)
      acc[n] = (v8f){0.f, 0.f, 0.f, 0.f, 0.f, 0.f, 0.f, 0.f};

#pragma unroll
    for (int kb = 0; kb < DE; kb += 32) {
      v16bf a = load_a_frag(sm_cat, SROW, mrow0 + lr, kb, lh);
#pragma unroll
      for (int n = 0; n < 4; ++n) {
        v16bf b = load_b_frag(sm_w1, SROW, n * 16 + lr, kb, lh);
        acc[n] = WMMA_BF16(a, b, acc[n]);
      }
    }
    // tanh + store bf16 (D layout: lane<16 -> M=vgpr, lane>=16 -> M=8+vgpr)
#pragma unroll
    for (int n = 0; n < 4; ++n)
#pragma unroll
      for (int i = 0; i < 8; ++i) {
        int r = mrow0 + lh * 8 + i;
        int c = n * 16 + lr;
        sm_h1[r * SH + c] = (__bf16)tanhf(acc[n][i]);
      }
  }
  __syncthreads();

  // ---- GEMM2: neib_att = h1[128x64] @ att2_w2.T[64x64], then ws straight
  //      from accumulators: each wave's 16-row tile belongs to node wave>>1.
  {
    v8f acc[4];
#pragma unroll
    for (int n = 0; n < 4; ++n)
      acc[n] = (v8f){0.f, 0.f, 0.f, 0.f, 0.f, 0.f, 0.f, 0.f};

#pragma unroll
    for (int kb = 0; kb < H_DIM; kb += 32) {
      v16bf a = load_a_frag(sm_h1, SH, mrow0 + lr, kb, lh);
#pragma unroll
      for (int n = 0; n < 4; ++n) {
        v16bf b = load_b_frag(sm_w2, SH, n * 16 + lr, kb, lh);
        acc[n] = WMMA_BF16(a, b, acc[n]);
      }
    }

    const int nw = wave >> 1;                 // node of this wave's rows
    float part[8];
#pragma unroll
    for (int i = 0; i < 8; ++i) {
      float s = 0.f;
#pragma unroll
      for (int n = 0; n < 4; ++n)
        s += acc[n][i] * sm_xatt[nw * 64 + n * 16 + lr];
      part[i] = s;
    }
    // reduce over the 16 lanes sharing each row (xor 1,2,4,8 stays in half)
#pragma unroll
    for (int i = 0; i < 8; ++i) {
#pragma unroll
      for (int m = 1; m < 16; m <<= 1)
        part[i] += __shfl_xor(part[i], m, 32);
    }
    if (lr == 0) {
#pragma unroll
      for (int i = 0; i < 8; ++i) {
        int k = (wave & 1) * 16 + lh * 8 + i;   // row index within node
        sm_ws[nw * 32 + k] = part[i];
      }
    }
  }
  __syncthreads();

  // ---- scale + mask + softmax over k (4 nodes, 1 thread each) -------------
  if (t < 4) {
    float* w = sm_ws + t * 32;
    const float* mr = sm_mask + t * 32;
    float mx = -1e30f;
    for (int k = 0; k < 32; ++k) {
      w[k] = w[k] * 0.125f + (-9999999.0f) * mr[k];
      mx = fmaxf(mx, w[k]);
    }
    float s = 0.f;
    for (int k = 0; k < 32; ++k) { float e = __expf(w[k] - mx); w[k] = e; s += e; }
    float inv = 1.f / s;
    for (int k = 0; k < 32; ++k) w[k] *= inv;
  }
  __syncthreads();

  // ---- agg[n][d] = sum_k ws[k] * cat[n*32+k][d] ---------------------------
  for (int i = t; i < 4 * DE; i += 256) {
    int n = i / DE, d = i % DE;
    const float*  p = sm_ws + n * 32;
    const __bf16* c = sm_cat + (n * 32) * SROW + d;
    float acc = 0.f;
#pragma unroll
    for (int k = 0; k < K_NB; ++k) acc += p[k] * (float)c[k * SROW];
    agg_out[(long)(node0 + n) * DE + d] = acc;
  }
}

// ---------------------------------------------------------------------------
// Kernel 2: out = relu([x @ fcx_w.T | agg @ fcn_w.T])  -> [N, 512] fp32
// WG = 256 threads computes a 128(M) x 64(N) tile; bf16 WMMA, staged via LDS.
// ---------------------------------------------------------------------------
#define SA 40   // 32 + 8 pad (bf16 elems)

__global__ __launch_bounds__(256)
void out_gemm_kernel(const float* __restrict__ x,
                     const float* __restrict__ agg,
                     const float* __restrict__ fcx_w,
                     const float* __restrict__ fcn_w,
                     float* __restrict__ out) {
  __shared__ __bf16 sA[128 * SA];   // A tile: 128 rows x 32 k
  __shared__ __bf16 sB[64 * SA];    // W tile: 64 out-cols x 32 k

  const int t     = threadIdx.x;
  const int nt_id = blockIdx.x & 7;    // 8 tiles of 64 output cols
  const int mt_id = blockIdx.x >> 3;   // 128 tiles of 128 rows
  const bool second = nt_id >= 4;
  const float* A  = second ? agg : x;
  const float* W  = second ? fcn_w : fcx_w;
  const int Kdim  = second ? DE : D_IN;
  const int wrow0 = (nt_id & 3) * 64;  // weight-matrix row (== output col in half)
  const int row0  = mt_id * 128;

  const int wave = t >> 5, lane = t & 31, lr = lane & 15, lh = lane >> 4;

  v8f acc[4];
#pragma unroll
  for (int n = 0; n < 4; ++n)
    acc[n] = (v8f){0.f, 0.f, 0.f, 0.f, 0.f, 0.f, 0.f, 0.f};

  for (int kb = 0; kb < Kdim; kb += 32) {
    for (int i = t; i < 128 * 8; i += 256) {     // A: 128 rows x 8 float4
      int r = i >> 3, c4 = i & 7;
      float4 v = *(const float4*)(A + (long)(row0 + r) * Kdim + kb + c4 * 4);
      st4bf(sA + r * SA + c4 * 4, v);
    }
    for (int i = t; i < 64 * 8; i += 256) {      // W: 64 rows x 8 float4
      int r = i >> 3, c4 = i & 7;
      float4 v = *(const float4*)(W + (long)(wrow0 + r) * Kdim + kb + c4 * 4);
      st4bf(sB + r * SA + c4 * 4, v);
    }
    __syncthreads();
    v16bf a = load_a_frag(sA, SA, wave * 16 + lr, 0, lh);
#pragma unroll
    for (int n = 0; n < 4; ++n) {
      v16bf b = load_b_frag(sB, SA, n * 16 + lr, 0, lh);
      acc[n] = WMMA_BF16(a, b, acc[n]);
    }
    __syncthreads();
  }

  // ReLU + store
  const int orow0 = row0 + wave * 16;
  const int ocol0 = nt_id * 64;
#pragma unroll
  for (int n = 0; n < 4; ++n)
#pragma unroll
    for (int i = 0; i < 8; ++i) {
      int r = orow0 + lh * 8 + i;
      int c = ocol0 + n * 16 + lr;
      out[(long)r * 512 + c] = fmaxf(acc[n][i], 0.f);
    }
}

// ---------------------------------------------------------------------------
// Launch
// ---------------------------------------------------------------------------
extern "C" void kernel_launch(void* const* d_in, const int* in_sizes, int n_in,
                              void* d_out, int out_size, void* d_ws, size_t ws_size,
                              hipStream_t stream) {
  const float* x       = (const float*)d_in[0];
  const float* neibs   = (const float*)d_in[1];
  const float* edge    = (const float*)d_in[2];
  const float* mask    = (const float*)d_in[3];
  const float* att_w1  = (const float*)d_in[4];
  const float* att_w2  = (const float*)d_in[5];
  const float* att2_w1 = (const float*)d_in[6];
  const float* att2_w2 = (const float*)d_in[7];
  const float* fcx_w   = (const float*)d_in[8];
  const float* fcn_w   = (const float*)d_in[9];
  float* agg = (float*)d_ws;          // N x 320 fp32 scratch (~21 MB)
  float* out = (float*)d_out;

  constexpr size_t SMEM1 =
      (128 * SROW + 64 * SROW + 64 * SH + 128 * SH) * sizeof(__bf16) +
      (4 * 256 + 4 * 64 + 4 * 64 + 4 * 32 + 4 * 32) * sizeof(float);  // ~157 KB -> 2 WG/WGP

  (void)hipFuncSetAttribute((const void*)attn_fused_kernel,
                            hipFuncAttributeMaxDynamicSharedMemorySize, (int)SMEM1);

  attn_fused_kernel<<<N_NODES / 4, 256, SMEM1, stream>>>(
      x, neibs, edge, mask, att_w1, att_w2, att2_w1, att2_w2, agg);

  out_gemm_kernel<<<(N_NODES / 128) * 8, 256, 0, stream>>>(x, agg, fcx_w, fcn_w, out);
}
